// ButterflyLinear_76862734729551
// MI455X (gfx1250) — matrix operand order repaired and checked by reading, not compiled, not run on hardware
//
#include <hip/hip_runtime.h>

// ---------------------------------------------------------------------------
// ButterflyLinear fused as a single dense GEMM:
//   out = x @ W_eff^T, W_eff folded from (stage1, shuffle, mix_w, stage2).
// GEMM: bf16x3 split precision on v_wmma_f32_16x16x32_bf16 (fp32 accum),
// double-buffered LDS, async global->LDS copies for the pre-split W tiles.
// x split uses truncation-hi + exact fp32 residual (same accuracy, far fewer
// VALU ops; packing via v_perm_b32).
// ---------------------------------------------------------------------------

typedef __attribute__((ext_vector_type(16))) __bf16 v16bf;
typedef __attribute__((ext_vector_type(8)))  __bf16 v8bf;
typedef __attribute__((ext_vector_type(8)))  float  v8f;
typedef int v4i_ __attribute__((vector_size(16)));

#define KDIM   2048
#define OUT_F  2048
#define NROWS  16384
#define BLK_M  128
#define BLK_N  128
#define BLK_K  32
#define NTILES (KDIM / BLK_K)
#define LDS_STRIDE 40   // 32 + 8 pad halves -> 80B row stride, conflict-friendly

__device__ __forceinline__ unsigned short f32_to_bf16_rne(float x) {
  unsigned u = __float_as_uint(x);
  unsigned r = u + 0x7FFFu + ((u >> 16) & 1u);
  return (unsigned short)(r >> 16);
}
__device__ __forceinline__ float bf16_bits_to_f32(unsigned short h) {
  return __uint_as_float(((unsigned)h) << 16);
}

// CDNA5 async global->LDS 16B copy (ASYNCcnt-tracked).
__device__ __forceinline__ void async_copy_b128(const void* g, void* l) {
#if defined(__has_builtin) && __has_builtin(__builtin_amdgcn_global_load_async_to_lds_b128)
  typedef __attribute__((address_space(1))) v4i_* gp_t;
  typedef __attribute__((address_space(3))) v4i_* lp_t;
  __builtin_amdgcn_global_load_async_to_lds_b128((gp_t)g, (lp_t)l, 0, 0);
#else
  unsigned lo = (unsigned)(unsigned long long)l;   // low 32 bits = LDS offset
  asm volatile("global_load_async_to_lds_b128 %0, %1, off"
               :: "v"(lo), "v"(g) : "memory");
#endif
}
__device__ __forceinline__ void wait_async0() {
  asm volatile("s_wait_asynccnt 0x0" ::: "memory");
}

// ---------------------------------------------------------------------------
// fold1: W1[o, g*64+c] = sum_d mix_w[o, (d%32)*64 + 2g + d/32] * stage1[g*64+c, d]
// (shuffle permutation absorbed into the mix_w column index)
// ---------------------------------------------------------------------------
__global__ void fold1_kernel(const float* __restrict__ mix_w,
                             const float* __restrict__ stage1,
                             float* __restrict__ W1) {
  int idx = blockIdx.x * blockDim.x + threadIdx.x;   // 2048*2048 elements
  int o = idx >> 11;
  int i = idx & 2047;
  int g = i >> 6;
  int c = i & 63;
  const float* mrow = mix_w + (size_t)o * KDIM;
  const float* srow = stage1 + (size_t)(g * 64 + c) * 64;
  float acc = 0.f;
#pragma unroll 8
  for (int d = 0; d < 64; ++d) {
    float m = mrow[(d & 31) * 64 + 2 * g + (d >> 5)];
    acc = fmaf(m, srow[d], acc);
  }
  W1[idx] = acc;
}

// ---------------------------------------------------------------------------
// fold2: W[m, i] = sum_c2 stage2[(m/64)*64+c2, m%64] * W1[(m/64)*64+c2, i]
// then split fp32 -> bf16 hi/lo planes.
// ---------------------------------------------------------------------------
__global__ void fold2_kernel(const float* __restrict__ W1,
                             const float* __restrict__ stage2,
                             unsigned short* __restrict__ Whi,
                             unsigned short* __restrict__ Wlo) {
  int idx = blockIdx.x * blockDim.x + threadIdx.x;
  int m = idx >> 11;
  int i = idx & 2047;
  int G  = m >> 6;
  int d2 = m & 63;
  float acc = 0.f;
#pragma unroll 8
  for (int c2 = 0; c2 < 64; ++c2) {
    float s = stage2[(size_t)(G * 64 + c2) * 64 + d2];
    float w = W1[(size_t)(G * 64 + c2) * KDIM + i];
    acc = fmaf(s, w, acc);
  }
  unsigned short hb = f32_to_bf16_rne(acc);
  unsigned short lb = f32_to_bf16_rne(acc - bf16_bits_to_f32(hb));
  Whi[idx] = hb;
  Wlo[idx] = lb;
}

// ---------------------------------------------------------------------------
// GEMM helpers
// ---------------------------------------------------------------------------
struct APrefetch { float4 v[4]; };

__device__ __forceinline__ void load_A_tile(const float* __restrict__ X,
                                            int rbase, int kk, int tid,
                                            APrefetch& pf) {
#pragma unroll
  for (int p = 0; p < 4; ++p) {
    int lin = tid + p * 256;            // 0..1023 float4 units
    int row = lin >> 3;                 // 128 rows
    int kq  = (lin & 7) << 2;           // 0..28 step 4
    pf.v[p] = *(const float4*)(X + (size_t)(rbase + row) * KDIM + kk + kq);
  }
}

// Truncation-hi split:  hi = x with low 16 bits zeroed (bf16 by truncation),
// lo = bf16(x - hi_f) where the fp32 residual is exact. hi+lo covers ~16
// mantissa bits of x, matching the bf16x3 scheme's accuracy.
// Packing two bf16 halves = one v_perm_b32 (selector 0x07060302 picks the
// high 16 bits of two fp32 words).
__device__ __forceinline__ void store_A_tile(const APrefetch& pf, int tid,
                                             unsigned short* Ah,
                                             unsigned short* Al) {
#pragma unroll
  for (int p = 0; p < 4; ++p) {
    int lin = tid + p * 256;
    int row = lin >> 3;
    int kq  = (lin & 7) << 2;
    float xs[4] = {pf.v[p].x, pf.v[p].y, pf.v[p].z, pf.v[p].w};
    unsigned u[4], lu[4];
#pragma unroll
    for (int e = 0; e < 4; ++e) {
      u[e] = __float_as_uint(xs[e]);
      float hf = __uint_as_float(u[e] & 0xFFFF0000u);
      lu[e] = __float_as_uint(xs[e] - hf);        // exact residual
    }
    unsigned hp0 = __builtin_amdgcn_perm(u[1],  u[0],  0x07060302u);
    unsigned hp1 = __builtin_amdgcn_perm(u[3],  u[2],  0x07060302u);
    unsigned lp0 = __builtin_amdgcn_perm(lu[1], lu[0], 0x07060302u);
    unsigned lp1 = __builtin_amdgcn_perm(lu[3], lu[2], 0x07060302u);
    *(uint2*)(Ah + row * LDS_STRIDE + kq) = make_uint2(hp0, hp1);
    *(uint2*)(Al + row * LDS_STRIDE + kq) = make_uint2(lp0, lp1);
  }
}

__device__ __forceinline__ void async_B_tile(const unsigned short* __restrict__ Whi,
                                             const unsigned short* __restrict__ Wlo,
                                             int nbase, int kk, int tid,
                                             unsigned short* Bh,
                                             unsigned short* Bl) {
  int row   = tid >> 1;                 // 128 rows
  int chunk = (tid & 1) << 3;           // element offset 0 or 8 (16B chunks)
  async_copy_b128(Whi + (size_t)(nbase + row) * KDIM + kk + chunk,
                  Bh + row * LDS_STRIDE + chunk);
  async_copy_b128(Wlo + (size_t)(nbase + row) * KDIM + kk + chunk,
                  Bl + row * LDS_STRIDE + chunk);
}

// ---------------------------------------------------------------------------
// Main GEMM: out[N=16384, M=2048] = X * W^T   (bf16x3 split, fp32 accumulate)
// 128x128 tile per workgroup, 8 wave32: wave (wm in 0..1, wn in 0..3) owns a
// 64x32 sub-tile = 4x2 WMMA tiles of 16x16. Double-buffered LDS pipeline.
// ---------------------------------------------------------------------------
__global__ __launch_bounds__(256)
void gemm_kernel(const float* __restrict__ X,
                 const unsigned short* __restrict__ Whi,
                 const unsigned short* __restrict__ Wlo,
                 float* __restrict__ out) {
  __shared__ __align__(16) unsigned short Ah[2][BLK_M * LDS_STRIDE];
  __shared__ __align__(16) unsigned short Al[2][BLK_M * LDS_STRIDE];
  __shared__ __align__(16) unsigned short Bh[2][BLK_N * LDS_STRIDE];
  __shared__ __align__(16) unsigned short Bl[2][BLK_N * LDS_STRIDE];

  const int tid   = threadIdx.x;
  const int rbase = blockIdx.x * BLK_M;   // x rows
  const int nbase = blockIdx.y * BLK_N;   // out features

  const int wave  = tid >> 5;
  const int lane  = tid & 31;
  const int wm    = wave >> 2;            // 0..1
  const int wn    = wave & 3;             // 0..3
  const int lhalf = lane >> 4;            // 0/1 (lane half)
  const int l15   = lane & 15;

  v8f acc[4][2];
#pragma unroll
  for (int mi = 0; mi < 4; ++mi)
#pragma unroll
    for (int ni = 0; ni < 2; ++ni)
#pragma unroll
      for (int e = 0; e < 8; ++e) acc[mi][ni][e] = 0.f;

  // ---- prologue: stage tile 0 into buffer 0 ----
  APrefetch pf;
  load_A_tile(X, rbase, 0, tid, pf);
  store_A_tile(pf, tid, Ah[0], Al[0]);
  async_B_tile(Whi, Wlo, nbase, 0, tid, Bh[0], Bl[0]);
  wait_async0();
  __syncthreads();

  for (int t = 0; t < NTILES; ++t) {
    const int  cur      = t & 1;
    const int  nxt      = cur ^ 1;
    const bool has_next = (t + 1) < NTILES;

    // issue next tile's global traffic early (hidden under the WMMA block)
    if (has_next) {
      load_A_tile(X, rbase, (t + 1) * BLK_K, tid, pf);
      async_B_tile(Whi, Wlo, nbase, (t + 1) * BLK_K, tid, Bh[nxt], Bl[nxt]);
    }

    // ---- fragment loads from buf[cur] (ISA 16-bit A/B VGPR layouts) ----
    union FragU { v16bf v; v8bf h[2]; };
    v16bf afh[4], afl[4], bfh[2], bfl[2];

#pragma unroll
    for (int mi = 0; mi < 4; ++mi) {
      int row = wm * 64 + mi * 16 + l15;
      int k0  = lhalf ? 8 : 0;            // lanes 0-15: K 0..7,16..23 ; 16-31: K 8..15,24..31
      FragU fh, fl;
      fh.h[0] = *(const v8bf*)(Ah[cur] + row * LDS_STRIDE + k0);
      fh.h[1] = *(const v8bf*)(Ah[cur] + row * LDS_STRIDE + k0 + 16);
      fl.h[0] = *(const v8bf*)(Al[cur] + row * LDS_STRIDE + k0);
      fl.h[1] = *(const v8bf*)(Al[cur] + row * LDS_STRIDE + k0 + 16);
      afh[mi] = fh.v; afl[mi] = fl.v;
    }
#pragma unroll
    for (int ni = 0; ni < 2; ++ni) {
      int col = wn * 32 + ni * 16 + l15;
      int k0  = lhalf ? 8 : 0;            // lanes 0-15: K 0..7,16..23 ; 16-31: K 8..15,24..31
      FragU fh, fl;
      fh.h[0] = *(const v8bf*)(Bh[cur] + col * LDS_STRIDE + k0);
      fh.h[1] = *(const v8bf*)(Bh[cur] + col * LDS_STRIDE + k0 + 16);
      fl.h[0] = *(const v8bf*)(Bl[cur] + col * LDS_STRIDE + k0);
      fl.h[1] = *(const v8bf*)(Bl[cur] + col * LDS_STRIDE + k0 + 16);
      bfh[ni] = fh.v; bfl[ni] = fl.v;
    }

    // ---- bf16x3 split-product WMMA ----
#pragma unroll
    for (int mi = 0; mi < 4; ++mi)
#pragma unroll
      for (int ni = 0; ni < 2; ++ni) {
        acc[mi][ni] = __builtin_amdgcn_wmma_f32_16x16x32_bf16(
            false, afh[mi], false, bfh[ni], (short)0, acc[mi][ni], false, false);
        acc[mi][ni] = __builtin_amdgcn_wmma_f32_16x16x32_bf16(
            false, afh[mi], false, bfl[ni], (short)0, acc[mi][ni], false, false);
        acc[mi][ni] = __builtin_amdgcn_wmma_f32_16x16x32_bf16(
            false, afl[mi], false, bfh[ni], (short)0, acc[mi][ni], false, false);
      }

    // convert + store next A tile (its global loads have landed by now)
    if (has_next) store_A_tile(pf, tid, Ah[nxt], Al[nxt]);

    wait_async0();       // B tile for buf[nxt] resident in LDS
    __syncthreads();     // all waves done reading buf[cur], writes visible
  }

  // ---- epilogue: C/D layout -> global (VGPR j: lanes 0-15 M=j, 16-31 M=j+8)
#pragma unroll
  for (int mi = 0; mi < 4; ++mi) {
#pragma unroll
    for (int ni = 0; ni < 2; ++ni) {
      int col = nbase + wn * 32 + ni * 16 + l15;
#pragma unroll
      for (int j = 0; j < 8; ++j) {
        int row = rbase + wm * 64 + mi * 16 + j + lhalf * 8;
        out[(size_t)row * OUT_F + col] = acc[mi][ni][j];
      }
    }
  }
}

// ---------------------------------------------------------------------------
extern "C" void kernel_launch(void* const* d_in, const int* in_sizes, int n_in,
                              void* d_out, int out_size, void* d_ws, size_t ws_size,
                              hipStream_t stream) {
  const float* x      = (const float*)d_in[0];   // [4,4096,2048] -> [16384,2048]
  const float* stage1 = (const float*)d_in[1];   // [2048,64]
  const float* stage2 = (const float*)d_in[2];   // [2048,64]
  const float* mix_w  = (const float*)d_in[3];   // [2048,2048]
  float* out = (float*)d_out;                    // [16384,2048]

  char* ws = (char*)d_ws;
  float*          W1  = (float*)ws;                          // 16 MB fp32
  unsigned short* Whi = (unsigned short*)(ws + (16u << 20)); //  8 MB bf16 hi
  unsigned short* Wlo = (unsigned short*)(ws + (24u << 20)); //  8 MB bf16 lo

  fold1_kernel<<<(2048 * 2048) / 256, 256, 0, stream>>>(mix_w, stage1, W1);
  fold2_kernel<<<(2048 * 2048) / 256, 256, 0, stream>>>(W1, stage2, Whi, Wlo);

  dim3 grid(NROWS / BLK_M, OUT_F / BLK_N);   // 128 x 16 workgroups
  gemm_kernel<<<grid, 256, 0, stream>>>(x, Whi, Wlo, out);
}
